// FSDMLBBoxHead_15625091022925
// MI455X (gfx1250) — matrix operand order repaired and checked by reading, not compiled, not run on hardware
//
#include <hip/hip_runtime.h>
#include <hip/hip_bf16.h>
#include <stdint.h>

// Problem constants (from reference)
#define DD    12544            // K (flattened RoI feature dim)
#define BB    4096             // M (batch of RoIs)
#define NCLS  80               // cls columns
#define NREG  324              // reg columns (4*81)
#define NCOMB 416              // 80 + 324 + 12 zero-pad  -> 26 N-tiles of 16
#define KC    32               // K-chunk == one bf16 WMMA K-step
#define NCH   (DD / KC)        // 392 chunks
#define MROWS 32               // rows per workgroup (128 WGs -> x read exactly once)
#define APW   13               // async B b128 instrs per wave per chunk
#define AIPW  14               // total async instrs per wave per chunk (13 B + 1 A)

typedef __attribute__((ext_vector_type(16))) __bf16 bf16x16;
typedef __attribute__((ext_vector_type(8)))  float  f32x8;

union FragB16 { bf16x16 v; uint4 q[2]; };
union F4 { float4 v; float f[4]; };

// ---------------------------------------------------------------------------
// Prepass: combined weight matrix [416 x 12544] as bf16 hi/lo split in d_ws
// (rows 0..79 = w_cls, 80..403 = w_reg, 404..415 = zeros) + per-class ||w||^2.
// ---------------------------------------------------------------------------
__global__ __launch_bounds__(256)
void fsdml_prep(const float* __restrict__ w_cls,
                const float* __restrict__ w_reg,
                __bf16* __restrict__ whi,
                __bf16* __restrict__ wlo,
                float*  __restrict__ colss) {
  const int c = blockIdx.x;                 // combined column 0..415
  const float* src = nullptr;
  if (c < NCLS)             src = w_cls + (size_t)c * DD;
  else if (c < NCLS + NREG) src = w_reg + (size_t)(c - NCLS) * DD;

  float ss = 0.0f;
  for (int k = threadIdx.x; k < DD; k += blockDim.x) {
    float v = src ? src[k] : 0.0f;
    __bf16 h = (__bf16)v;
    __bf16 l = (__bf16)(v - (float)h);      // residual for bf16x2 split
    whi[(size_t)c * DD + k] = h;
    wlo[(size_t)c * DD + k] = l;
    ss += v * v;
  }
  __shared__ float red[256];
  red[threadIdx.x] = ss;
  __syncthreads();
  for (int s = 128; s > 0; s >>= 1) {
    if ((int)threadIdx.x < s) red[threadIdx.x] += red[threadIdx.x + s];
    __syncthreads();
  }
  if (threadIdx.x == 0) colss[c] = red[0];
}

// ---------------------------------------------------------------------------
// Main fused kernel: 128 WGs x 256 threads (8 wave32).
// ALL global->LDS movement is async (ASYNCcnt, double buffered):
//   - B: bf16 hi/lo weight tiles from d_ws
//   - A: raw fp32 x tile; fp32->(bf16 hi, lo) split done in registers while
//     building A fragments (so barriers never drain LOADcnt)
// 3-product bf16x2 WMMA accumulation in f32; fused cls/reg epilogue.
// ---------------------------------------------------------------------------
__global__ __launch_bounds__(256)
void fsdml_main(const float* __restrict__ x,
                const float* __restrict__ b_reg,
                const __bf16* __restrict__ whi,
                const __bf16* __restrict__ wlo,
                const float* __restrict__ colss,
                float* __restrict__ out) {
  // LDS: 106496 (B) + 8192 (A fp32) + 128 = 114816 B
  __shared__ __bf16 sB[2][2][NCOMB * KC];   // [buf][hi/lo][col*KC + k]
  __shared__ float  sX[2][MROWS * KC];      // [buf][row*KC + k] raw fp32 A
  __shared__ float  rowss[MROWS];

  const int tid  = threadIdx.x;
  const int lane = tid & 31;
  const int wave = tid >> 5;

  if (tid < MROWS) rowss[tid] = 0.0f;

  // wave -> (M-tile, N-tile range): 2 M-tiles x 26 N-tiles over 8 waves
  const int mbase  = (wave & 1) * 16;
  const int nh     = wave >> 1;                // 0..3
  const int nstart = (nh < 3) ? nh * 7 : 20;   // 0,7,14,20
  const int ncnt   = (nh < 2) ? 7 : 6;         // 7+7+6+6 = 26

  f32x8 acc[7];
  #pragma unroll
  for (int j = 0; j < 7; ++j)
    #pragma unroll
    for (int e = 0; e < 8; ++e) acc[j][e] = 0.0f;

  float ss = 0.0f;

  // Issue this wave's share of the async copies for one chunk:
  // 13 instrs of B (hi/lo planes; plane uniform per instr, boundary at j==52)
  // + 1 instr of A (fp32 tile).
  auto issueChunk = [&](int buf, int k0) {
    unsigned lbaseB = (unsigned)(uintptr_t)(&sB[buf][0][0]);
    #pragma unroll
    for (int i = 0; i < APW; ++i) {
      int j     = wave * APW + i;              // 0..103
      int plane = (j >= 52) ? 1 : 0;
      int t     = (j - plane * 52) * 32 + lane;   // 16B transfer id, 0..1663
      int c     = t >> 2;                      // combined column
      int kq    = t & 3;                       // which 16B of this K-chunk row
      unsigned goff = (unsigned)((c * DD + k0 + kq * 8) * 2);
      unsigned loff = lbaseB + (unsigned)(plane * (NCOMB * KC * 2) + c * (KC * 2) + kq * 16);
      unsigned long long gb = (unsigned long long)(uintptr_t)(plane ? wlo : whi);
      asm volatile("global_load_async_to_lds_b128 %0, %1, %2"
                   :: "v"(loff), "v"(goff), "s"(gb) : "memory");
    }
    {
      unsigned lbaseX = (unsigned)(uintptr_t)(&sX[buf][0]);
      int t   = wave * 32 + lane;              // 0..255 16B transfers
      int row = t >> 3;
      int kq  = t & 7;                         // 8 x float4 per 32-float row
      unsigned goff = (unsigned)((((unsigned)blockIdx.x * MROWS + row) * DD + k0 + kq * 4) * 4);
      unsigned loff = lbaseX + (unsigned)(row * (KC * 4) + kq * 16);
      unsigned long long gb = (unsigned long long)(uintptr_t)x;
      asm volatile("global_load_async_to_lds_b128 %0, %1, %2"
                   :: "v"(loff), "v"(goff), "s"(gb) : "memory");
    }
  };

  // Pipeline prologue
  issueChunk(0, 0);

  const int rsel  = lane & 15;
  const int aksel = (lane >> 4) * 16;  // A k order used here (contiguous): lanes0-15 K0-15, lanes16-31 K16-31
  const int bksel = (lane >> 4) * 16;  // B k order used here (same as A): lanes0-15 K0-15, lanes16-31 K16-31

  for (int cc = 0; cc < NCH; ++cc) {
    const int cur = cc & 1;

    if (cc + 1 < NCH) {
      issueChunk(cur ^ 1, (cc + 1) * KC);     // prefetch next chunk (other buffer)
      asm volatile("s_wait_asynccnt 14" ::: "memory");  // chunk cc is in LDS
    } else {
      asm volatile("s_wait_asynccnt 0" ::: "memory");
    }
    __syncthreads();

    // --- build A fragments: fp32 from LDS -> bf16 hi + residual lo ---------
    FragB16 ahi, alo;
    {
      const float* px = &sX[cur][(mbase + rsel) * KC + aksel];
      F4 xa, xb, xc, xd;
      xa.v = *(const float4*)(px);        // K aksel..aksel+3
      xb.v = *(const float4*)(px + 4);    // K aksel+4..aksel+7
      xc.v = *(const float4*)(px + 8);    // K aksel+8..
      xd.v = *(const float4*)(px + 12);
      float f[16];
      #pragma unroll
      for (int e = 0; e < 4; ++e) {
        f[e]      = xa.f[e];
        f[4 + e]  = xb.f[e];
        f[8 + e]  = xc.f[e];
        f[12 + e] = xd.f[e];
      }
      #pragma unroll
      for (int e = 0; e < 16; ++e) {
        __bf16 h = (__bf16)f[e];
        ahi.v[e] = h;
        alo.v[e] = (__bf16)(f[e] - (float)h);
      }
      if (nh == 0) {                     // waves 0,1 cover every element once:
        float p = 0.0f;                  // exact fp32 row sum-of-squares
        #pragma unroll
        for (int e = 0; e < 16; ++e) p += f[e] * f[e];
        ss += p;
      }
    }

    // --- B fragments + WMMA ------------------------------------------------
    #pragma unroll
    for (int j = 0; j < 7; ++j) {
      if (j >= ncnt) break;
      const int col0 = (nstart + j) * 16;
      FragB16 bhi, blo;
      bhi.v = *(const bf16x16*)&sB[cur][0][(col0 + rsel) * KC + bksel];
      blo.v = *(const bf16x16*)&sB[cur][1][(col0 + rsel) * KC + bksel];
      // bf16x2 split: hi*hi + hi*lo + lo*hi  (lo*lo below fp32 rounding)
      acc[j] = __builtin_amdgcn_wmma_f32_16x16x32_bf16(false, ahi.v, false, bhi.v,
                                                       (short)0, acc[j], false, false);
      acc[j] = __builtin_amdgcn_wmma_f32_16x16x32_bf16(false, ahi.v, false, blo.v,
                                                       (short)0, acc[j], false, false);
      acc[j] = __builtin_amdgcn_wmma_f32_16x16x32_bf16(false, alo.v, false, bhi.v,
                                                       (short)0, acc[j], false, false);
    }
    __syncthreads();                          // protect both buffers before reuse
  }

  // --- epilogue ------------------------------------------------------------
  if (nh == 0) atomicAdd(&rowss[mbase + rsel], ss);   // 2 partials/row -> ds_add_f32
  __syncthreads();

  const int mhi = (lane >> 4) * 8;            // C/D layout: lanes16-31 hold M+8
  #pragma unroll
  for (int j = 0; j < 7; ++j) {
    if (j >= ncnt) break;
    const int col = (nstart + j) * 16 + rsel;
    #pragma unroll
    for (int r = 0; r < 8; ++r) {
      const int rl   = mbase + mhi + r;
      const int grow = blockIdx.x * MROWS + rl;
      const float v  = acc[j][r];
      if (col < NCLS) {
        float nr   = fmaxf(sqrtf(rowss[rl]), 1e-12f);   // F.normalize eps
        float nc   = fmaxf(sqrtf(colss[col]), 1e-12f);
        float cosv = v / (nr * nc);
        float d2   = fmaxf(2.0f - 2.0f * cosv, 0.0f);
        float s    = __expf(-2.0f * d2);                // exp(-d2/(2*0.25))
        float x1   = fmaxf(s, 1e-5f);                   // inverse_sigmoid
        float x2   = fmaxf(1.0f - s, 1e-5f);
        out[(size_t)grow * NCLS + col] = __logf(x1 / x2);
      } else if (col < NCLS + NREG) {
        const int rc = col - NCLS;
        out[(size_t)BB * NCLS + (size_t)grow * NREG + rc] = v + b_reg[rc];
      } // cols 404..415 are padding: dropped
    }
  }
}

// ---------------------------------------------------------------------------
extern "C" void kernel_launch(void* const* d_in, const int* in_sizes, int n_in,
                              void* d_out, int out_size, void* d_ws, size_t ws_size,
                              hipStream_t stream) {
  const float* x     = (const float*)d_in[0];
  const float* w_cls = (const float*)d_in[1];
  const float* w_reg = (const float*)d_in[2];
  const float* b_reg = (const float*)d_in[3];
  float* out = (float*)d_out;

  // d_ws layout: whi[416*12544] bf16 | wlo[416*12544] bf16 | colss[416] f32
  __bf16* whi   = (__bf16*)d_ws;
  __bf16* wlo   = whi + (size_t)NCOMB * DD;
  float*  colss = (float*)(wlo + (size_t)NCOMB * DD);   // ~20.9 MB total

  fsdml_prep<<<NCOMB, 256, 0, stream>>>(w_cls, w_reg, whi, wlo, colss);
  fsdml_main<<<BB / MROWS, 256, 0, stream>>>(x, b_reg, whi, wlo, colss, out);
}
